// GatedLTMMemory_89489938580139
// MI455X (gfx1250) — matrix operand order repaired and checked by reading, not compiled, not run on hardware
//
#include <hip/hip_runtime.h>
#include <hip/hip_bf16.h>

// ---------------------------------------------------------------------------
// Types for gfx1250 WMMA (wave32).
// ---------------------------------------------------------------------------
typedef __attribute__((ext_vector_type(16))) __bf16 v16bf;
typedef __attribute__((ext_vector_type(8)))  __bf16 v8bf;
typedef __attribute__((ext_vector_type(8)))  float  v8f;

// Problem constants (match the reference).
#define BB 4
#define NN 1024
#define QDIM 320
#define DDIM 512
#define SS 1024
#define HH 8
#define KK 32
#define DH 64
#define MTOK (BB * NN)   // 4096 tokens

// ---------------------------------------------------------------------------
// Fragment loaders following CDNA5 ISA §7.12.2 VGPR layouts.
// A (16x32 bf16): lanes 0-15 = M rows with K {0..7,16..23}; lanes 16-31 same
// rows with K {8..15,24..31}.  Two contiguous 16-byte chunks per lane.
// ---------------------------------------------------------------------------
__device__ __forceinline__ v16bf load_frag_a(const __bf16* __restrict__ rowp,
                                             int k0, int hi) {
  v8bf lo = *reinterpret_cast<const v8bf*>(rowp + k0 + hi * 8);
  v8bf hv = *reinterpret_cast<const v8bf*>(rowp + k0 + 16 + hi * 8);
  return __builtin_shufflevector(lo, hv, 0, 1, 2, 3, 4, 5, 6, 7,
                                 8, 9, 10, 11, 12, 13, 14, 15);
}

// B (32x16 bf16): lanes 0-15 = columns with K {0..7,16..23}; lanes 16-31 K {8..15,24..31}.
// B is stored as [N,K] row-major (we compute A·Bᵀ), so per-lane data is
// two 16-byte chunks.
__device__ __forceinline__ v16bf load_frag_b(const __bf16* __restrict__ colp,
                                             int k0, int hi) {
  v8bf lo = *reinterpret_cast<const v8bf*>(colp + k0 + hi * 8);
  v8bf hv = *reinterpret_cast<const v8bf*>(colp + k0 + 16 + hi * 8);
  return __builtin_shufflevector(lo, hv, 0, 1, 2, 3, 4, 5, 6, 7,
                                 8, 9, 10, 11, 12, 13, 14, 15);
}

#define WMMA_BF16(A_, B_, C_) \
  __builtin_amdgcn_wmma_f32_16x16x32_bf16(false, (A_), false, (B_), (short)0, (C_), false, false)

// ---------------------------------------------------------------------------
// Generic bf16 WMMA GEMM:  C[M,N] = A[M,K] · B[N,K]ᵀ (+ bias[N]).
// Block = 128 threads = 4 waves; each wave owns a 32x64 output tile:
// 2 A fragments x 4 B fragments -> 8 v_wmma per K-slab of 32.
// Software-pipelined: K-slab k loads are issued before the WMMAs of slab
// k-32 execute, so vmem latency overlaps the XDL pipe instead of
// serializing on s_wait_loadcnt 0.  Grid (N/64, M/128).
// ---------------------------------------------------------------------------
template <typename OutT, bool HasBias>
__global__ __launch_bounds__(128)
void gemm_bf16_wmma_kernel(const __bf16* __restrict__ A,
                           const __bf16* __restrict__ B,
                           OutT* __restrict__ C,
                           const float* __restrict__ bias,
                           int M, int N, int Kd) {
  const int lane = threadIdx.x & 31;
  const int wave = threadIdx.x >> 5;
  const int row  = lane & 15;   // M row (A) / N col (B,C)
  const int hi   = lane >> 4;   // lane-group selector

  const int tileN = blockIdx.x * 64;
  const int tileM = blockIdx.y * 128 + wave * 32;

  const __bf16* __restrict__ A0 = A + (size_t)(tileM +  0 + row) * Kd;
  const __bf16* __restrict__ A1 = A + (size_t)(tileM + 16 + row) * Kd;
  const __bf16* __restrict__ B0 = B + (size_t)(tileN +  0 + row) * Kd;
  const __bf16* __restrict__ B1 = B + (size_t)(tileN + 16 + row) * Kd;
  const __bf16* __restrict__ B2 = B + (size_t)(tileN + 32 + row) * Kd;
  const __bf16* __restrict__ B3 = B + (size_t)(tileN + 48 + row) * Kd;

  v8f acc[8] = {};

  // Prologue: fragments for K-slab 0.
  v16bf a0 = load_frag_a(A0, 0, hi);
  v16bf a1 = load_frag_a(A1, 0, hi);
  v16bf b0 = load_frag_b(B0, 0, hi);
  v16bf b1 = load_frag_b(B1, 0, hi);
  v16bf b2 = load_frag_b(B2, 0, hi);
  v16bf b3 = load_frag_b(B3, 0, hi);

  // Pipelined main loop: issue slab-k loads, then compute slab-(k-32).
  for (int k0 = 32; k0 < Kd; k0 += 32) {
    __builtin_prefetch(A0 + k0 + 32, 0, 1);   // gfx1250 global_prefetch_b8

    v16bf a0n = load_frag_a(A0, k0, hi);
    v16bf a1n = load_frag_a(A1, k0, hi);
    v16bf b0n = load_frag_b(B0, k0, hi);
    v16bf b1n = load_frag_b(B1, k0, hi);
    v16bf b2n = load_frag_b(B2, k0, hi);
    v16bf b3n = load_frag_b(B3, k0, hi);

    acc[0] = WMMA_BF16(a0, b0, acc[0]);
    acc[1] = WMMA_BF16(a0, b1, acc[1]);
    acc[2] = WMMA_BF16(a0, b2, acc[2]);
    acc[3] = WMMA_BF16(a0, b3, acc[3]);
    acc[4] = WMMA_BF16(a1, b0, acc[4]);
    acc[5] = WMMA_BF16(a1, b1, acc[5]);
    acc[6] = WMMA_BF16(a1, b2, acc[6]);
    acc[7] = WMMA_BF16(a1, b3, acc[7]);

    a0 = a0n; a1 = a1n;
    b0 = b0n; b1 = b1n; b2 = b2n; b3 = b3n;
  }

  // Epilogue compute for the final K-slab.
  acc[0] = WMMA_BF16(a0, b0, acc[0]);
  acc[1] = WMMA_BF16(a0, b1, acc[1]);
  acc[2] = WMMA_BF16(a0, b2, acc[2]);
  acc[3] = WMMA_BF16(a0, b3, acc[3]);
  acc[4] = WMMA_BF16(a1, b0, acc[4]);
  acc[5] = WMMA_BF16(a1, b1, acc[5]);
  acc[6] = WMMA_BF16(a1, b2, acc[6]);
  acc[7] = WMMA_BF16(a1, b3, acc[7]);

  // C/D layout: VGPR r -> M = r + 8*hi (+16 for second A frag),
  // N = tileN + t*16 + (lane&15).
#pragma unroll
  for (int w2 = 0; w2 < 2; ++w2) {
#pragma unroll
    for (int t = 0; t < 4; ++t) {
      const v8f& a = acc[w2 * 4 + t];
      const int col = tileN + t * 16 + row;
      float bv = 0.0f;
      if (HasBias) bv = bias[col];
#pragma unroll
      for (int r = 0; r < 8; ++r) {
        const int m = tileM + w2 * 16 + hi * 8 + r;
        C[(size_t)m * N + col] = (OutT)(a[r] + bv);
      }
    }
  }
}

// ---------------------------------------------------------------------------
// Elementwise f32 -> bf16 conversion.
// ---------------------------------------------------------------------------
__global__ void f32_to_bf16_kernel(const float* __restrict__ in,
                                   __bf16* __restrict__ out, int n) {
  int i = blockIdx.x * blockDim.x + threadIdx.x;
  if (i < n) out[i] = (__bf16)in[i];
}

// ---------------------------------------------------------------------------
// Row-wise L2 normalize [R, D] -> bf16.  One wave per row.
// ---------------------------------------------------------------------------
__global__ void l2norm_bf16_kernel(const float* __restrict__ in,
                                   __bf16* __restrict__ out, int D) {
  const int row  = blockIdx.x;
  const int lane = threadIdx.x;
  const float* __restrict__ p = in + (size_t)row * D;
  float ss = 0.0f;
  for (int i = lane; i < D; i += 32) { float v = p[i]; ss += v * v; }
  for (int off = 16; off; off >>= 1) ss += __shfl_xor(ss, off, 32);
  const float inv = rsqrtf(ss + 1e-12f);
  for (int i = lane; i < D; i += 32)
    out[(size_t)row * D + i] = (__bf16)(p[i] * inv);
}

// ---------------------------------------------------------------------------
// Top-32 of S=1024 per row. One wave per row; 32 values/lane in registers,
// 32 rounds of shuffle arg-max (descending order, lower index wins ties).
// ---------------------------------------------------------------------------
__global__ void topk32_kernel(const float* __restrict__ scores,
                              int* __restrict__ idx) {
  const int row  = blockIdx.x;
  const int lane = threadIdx.x;
  const float* __restrict__ p = scores + (size_t)row * SS;
  float v[32];
#pragma unroll
  for (int j = 0; j < 32; ++j) v[j] = p[lane + j * 32];

  for (int t = 0; t < KK; ++t) {
    float best = -3.4e38f;
    int bj = 0;
#pragma unroll
    for (int j = 0; j < 32; ++j)
      if (v[j] > best) { best = v[j]; bj = j; }
    int bidx = lane + bj * 32;
    for (int off = 16; off; off >>= 1) {
      float ov = __shfl_xor(best, off, 32);
      int   oi = __shfl_xor(bidx, off, 32);
      if (ov > best || (ov == best && oi < bidx)) { best = ov; bidx = oi; }
    }
    if (lane == 0) idx[(size_t)row * KK + t] = bidx;
    if ((bidx & 31) == lane) v[bidx >> 5] = -3.4e38f;
  }
}

// ---------------------------------------------------------------------------
// Gather attention over pre-projected slots.  One wave per (token, head):
// lane k = slot k (dot over DH=64, softmax over 32 lanes), then each lane
// accumulates 2 output dims across all 32 slots via shuffles.
// ---------------------------------------------------------------------------
__global__ void attn_kernel(const float* __restrict__ qh,
                            const float* __restrict__ Kp,
                            const float* __restrict__ Vp,
                            const int* __restrict__ idx,
                            __bf16* __restrict__ ctx) {
  const int token = blockIdx.x;
  const int h     = blockIdx.y;
  const int lane  = threadIdx.x;

  const int slot = idx[(size_t)token * KK + lane];
  const float* __restrict__ qp = qh + (size_t)token * DDIM + h * DH;
  const float* __restrict__ kp = Kp + (size_t)slot  * DDIM + h * DH;

  float dot = 0.0f;
#pragma unroll
  for (int d = 0; d < DH; ++d) dot += qp[d] * kp[d];
  dot *= 0.125f;  // 1/sqrt(64)

  float m = dot;
  for (int off = 16; off; off >>= 1) m = fmaxf(m, __shfl_xor(m, off, 32));
  const float e = __expf(dot - m);
  float s = e;
  for (int off = 16; off; off >>= 1) s += __shfl_xor(s, off, 32);
  const float w = e / s;

  float c0 = 0.0f, c1 = 0.0f;
#pragma unroll
  for (int k = 0; k < KK; ++k) {
    const float wk = __shfl(w, k, 32);
    const int   sk = __shfl(slot, k, 32);
    const float* __restrict__ vp = Vp + (size_t)sk * DDIM + h * DH + lane * 2;
    c0 += wk * vp[0];
    c1 += wk * vp[1];
  }
  __bf16* __restrict__ op = ctx + (size_t)token * DDIM + h * DH + lane * 2;
  op[0] = (__bf16)c0;
  op[1] = (__bf16)c1;
}

// ---------------------------------------------------------------------------
// LayerNorm over D=512 per row -> bf16.  One wave per row, 16 elems/lane.
// ---------------------------------------------------------------------------
__global__ void layernorm_bf16_kernel(const float* __restrict__ in,
                                      const float* __restrict__ g,
                                      const float* __restrict__ b,
                                      __bf16* __restrict__ out) {
  const int row  = blockIdx.x;
  const int lane = threadIdx.x;
  const float* __restrict__ p = in + (size_t)row * DDIM;
  float vals[16];
  float sum = 0.0f, sq = 0.0f;
#pragma unroll
  for (int j = 0; j < 16; ++j) {
    float v = p[lane + j * 32];
    vals[j] = v; sum += v; sq += v * v;
  }
  for (int off = 16; off; off >>= 1) {
    sum += __shfl_xor(sum, off, 32);
    sq  += __shfl_xor(sq,  off, 32);
  }
  const float mu  = sum * (1.0f / DDIM);
  const float var = sq * (1.0f / DDIM) - mu * mu;
  const float inv = rsqrtf(var + 1e-5f);
#pragma unroll
  for (int j = 0; j < 16; ++j) {
    const int c = lane + j * 32;
    out[(size_t)row * DDIM + c] = (__bf16)((vals[j] - mu) * inv * g[c] + b[c]);
  }
}

// ---------------------------------------------------------------------------
// Host-side orchestration.
// ---------------------------------------------------------------------------
static inline void conv_bf16(const float* in, __bf16* out, int n, hipStream_t s) {
  f32_to_bf16_kernel<<<(n + 255) / 256, 256, 0, s>>>(in, out, n);
}

extern "C" void kernel_launch(void* const* d_in, const int* in_sizes, int n_in,
                              void* d_out, int out_size, void* d_ws, size_t ws_size,
                              hipStream_t stream) {
  (void)in_sizes; (void)n_in; (void)out_size; (void)ws_size;

  const float* query      = (const float*)d_in[0];   // [B,N,QD]
  const float* Wqp        = (const float*)d_in[1];   // [D,QD]
  const float* mem_keys   = (const float*)d_in[2];   // [S,D]
  const float* mem_values = (const float*)d_in[3];   // [S,D]
  const float* Wq         = (const float*)d_in[4];   // [D,D]
  const float* Wk         = (const float*)d_in[5];   // [D,D]
  const float* Wv         = (const float*)d_in[6];   // [D,D]
  const float* Wo         = (const float*)d_in[7];   // [D,D]
  const float* ln_g       = (const float*)d_in[8];   // [D]
  const float* ln_b       = (const float*)d_in[9];   // [D]
  const float* Wout       = (const float*)d_in[10];  // [QD,D]
  const float* bout       = (const float*)d_in[11];  // [QD]
  float* out = (float*)d_out;                        // [B,N,QD]

  // Bump allocator over workspace (256B aligned slabs).
  char* ws = (char*)d_ws;
  size_t off = 0;
  auto alloc = [&](size_t bytes) -> void* {
    void* p = ws + off;
    off += (bytes + 255) & ~(size_t)255;
    return p;
  };

  __bf16* queryB  = (__bf16*)alloc((size_t)MTOK * QDIM * 2);
  __bf16* WqpB    = (__bf16*)alloc((size_t)DDIM * QDIM * 2);
  __bf16* WqB     = (__bf16*)alloc((size_t)DDIM * DDIM * 2);
  __bf16* WkB     = (__bf16*)alloc((size_t)DDIM * DDIM * 2);
  __bf16* WvB     = (__bf16*)alloc((size_t)DDIM * DDIM * 2);
  __bf16* WoB     = (__bf16*)alloc((size_t)DDIM * DDIM * 2);
  __bf16* WoutB   = (__bf16*)alloc((size_t)QDIM * DDIM * 2);
  __bf16* keysNB  = (__bf16*)alloc((size_t)SS * DDIM * 2);
  __bf16* valsNB  = (__bf16*)alloc((size_t)SS * DDIM * 2);
  __bf16* qB      = (__bf16*)alloc((size_t)MTOK * DDIM * 2);
  float*  Kp      = (float*) alloc((size_t)SS * DDIM * 4);
  float*  Vp      = (float*) alloc((size_t)SS * DDIM * 4);
  float*  scores  = (float*) alloc((size_t)MTOK * SS * 4);
  int*    idx     = (int*)   alloc((size_t)MTOK * KK * 4);
  float*  qh      = (float*) alloc((size_t)MTOK * DDIM * 4);
  __bf16* ctxB    = (__bf16*)alloc((size_t)MTOK * DDIM * 2);
  float*  ctxo    = (float*) alloc((size_t)MTOK * DDIM * 4);
  __bf16* normedB = (__bf16*)alloc((size_t)MTOK * DDIM * 2);

  // ---- Stage 0: precision conversion of inputs -> bf16 ----
  conv_bf16(query, queryB, MTOK * QDIM, stream);
  conv_bf16(Wqp,   WqpB,   DDIM * QDIM, stream);
  conv_bf16(Wq,    WqB,    DDIM * DDIM, stream);
  conv_bf16(Wk,    WkB,    DDIM * DDIM, stream);
  conv_bf16(Wv,    WvB,    DDIM * DDIM, stream);
  conv_bf16(Wo,    WoB,    DDIM * DDIM, stream);
  conv_bf16(Wout,  WoutB,  QDIM * DDIM, stream);

  // ---- Stage 1: l2-normalize memory slots ----
  l2norm_bf16_kernel<<<SS, 32, 0, stream>>>(mem_keys,   keysNB, DDIM);
  l2norm_bf16_kernel<<<SS, 32, 0, stream>>>(mem_values, valsNB, DDIM);

  // ---- Stage 2: pre-project memory slots once (the big algebraic win) ----
  // Kp = keysN · Wkᵀ, Vp = valsN · Wvᵀ   (S x D)
  gemm_bf16_wmma_kernel<float, false><<<dim3(DDIM / 64, SS / 128), 128, 0, stream>>>(
      keysNB, WkB, Kp, nullptr, SS, DDIM, DDIM);
  gemm_bf16_wmma_kernel<float, false><<<dim3(DDIM / 64, SS / 128), 128, 0, stream>>>(
      valsNB, WvB, Vp, nullptr, SS, DDIM, DDIM);

  // ---- Stage 3: q = query · Wqpᵀ  (4096 x 512, K=320) ----
  gemm_bf16_wmma_kernel<__bf16, false><<<dim3(DDIM / 64, MTOK / 128), 128, 0, stream>>>(
      queryB, WqpB, qB, nullptr, MTOK, DDIM, QDIM);

  // ---- Stage 4: scores = q · keysNᵀ  (4096 x 1024) ----
  gemm_bf16_wmma_kernel<float, false><<<dim3(SS / 64, MTOK / 128), 128, 0, stream>>>(
      qB, keysNB, scores, nullptr, MTOK, SS, DDIM);

  // ---- Stage 5: top-32 slot preselection per token ----
  topk32_kernel<<<MTOK, 32, 0, stream>>>(scores, idx);

  // ---- Stage 6: qh = q · Wqᵀ ----
  gemm_bf16_wmma_kernel<float, false><<<dim3(DDIM / 64, MTOK / 128), 128, 0, stream>>>(
      qB, WqB, qh, nullptr, MTOK, DDIM, DDIM);

  // ---- Stage 7: gather attention over pre-projected slots -> ctx (bf16) ----
  attn_kernel<<<dim3(MTOK, HH), 32, 0, stream>>>(qh, Kp, Vp, idx, ctxB);

  // ---- Stage 8: ctxo = ctx · Woᵀ ----
  gemm_bf16_wmma_kernel<float, false><<<dim3(DDIM / 64, MTOK / 128), 128, 0, stream>>>(
      ctxB, WoB, ctxo, nullptr, MTOK, DDIM, DDIM);

  // ---- Stage 9: LayerNorm -> bf16 ----
  layernorm_bf16_kernel<<<MTOK, 32, 0, stream>>>(ctxo, ln_g, ln_b, normedB);

  // ---- Stage 10: out = normed · Woutᵀ + bout  (4096 x 320) ----
  gemm_bf16_wmma_kernel<float, true><<<dim3(QDIM / 64, MTOK / 128), 128, 0, stream>>>(
      normedB, WoutB, out, bout, MTOK, QDIM, DDIM);
}
